// MoE_25391846654148
// MI455X (gfx1250) — matrix operand order repaired and checked by reading, not compiled, not run on hardware
//
#include <hip/hip_runtime.h>
#include <stdint.h>
#include <stddef.h>

// Problem constants (B=4, S=2048 -> N=8192 tokens)
#define NTOK   8192
#define HDIM   1024
#define FDIM   4096
#define NEXP   8

#define TILE_M    64
#define FCHUNK    32
#define XS_STRIDE 1032   // bf16 units; 516 dwords/row, 516%64=4 -> conflict-free A reads
#define HS_STRIDE 40     // bf16 units; 20 dwords/row

typedef __attribute__((ext_vector_type(16))) __bf16 v16bf;
typedef __attribute__((ext_vector_type(8)))  float  v8f;

union Frag { v16bf f; uint32_t u[8]; };

__device__ __forceinline__ uint16_t f2bf(float x) {
  uint32_t u = __float_as_uint(x);
  u += 0x7FFFu + ((u >> 16) & 1u);   // round-to-nearest-even
  return (uint16_t)(u >> 16);
}

__device__ __forceinline__ v8f vzero8() {
  v8f z;
#pragma unroll
  for (int i = 0; i < 8; ++i) z[i] = 0.f;
  return z;
}

// A-matrix 16x32 bf16 fragment: lane<16 -> M=lane, K=0..15 in v0..7 (pairs);
// lane>=16 -> same M, K=16..31. rowptr points at row start + kbase.
__device__ __forceinline__ Frag load_a_frag(const uint16_t* rowptr, int half) {
  Frag a;
  const int base = half * 16;
#pragma unroll
  for (int v = 0; v < 4; ++v)
    a.u[v] = *(const uint32_t*)(rowptr + base + 2 * v);
#pragma unroll
  for (int v = 0; v < 4; ++v)
    a.u[4 + v] = *(const uint32_t*)(rowptr + 8 + base + 2 * v);
  return a;
}

// B-matrix 32x16 bf16 fragment: lane<16 -> N=lane, K=0..15 in v0..7 (pairs);
// lane>=16 -> N=lane-16, K=16..31. colptr points at column base + kbase + half*16
// in a layout where the K dimension is contiguous in memory.
__device__ __forceinline__ Frag load_b_frag(const uint16_t* colptr) {
  Frag b;
#pragma unroll
  for (int v = 0; v < 8; ++v)
    b.u[v] = *(const uint32_t*)(colptr + 2 * v);
  return b;
}

// CDNA5 async gather: MEM[g..g+15] -> LDS[lds_off..lds_off+15], tracked by ASYNCcnt.
__device__ __forceinline__ void async_gather_b128(uint32_t lds_off, const void* g) {
  asm volatile("global_load_async_to_lds_b128 %0, %1, off"
               :: "v"(lds_off), "v"(g) : "memory");
}
__device__ __forceinline__ void wait_asynccnt0() {
  asm volatile("s_wait_asynccnt 0x0" ::: "memory");
}

// ---------------------------------------------------------------- utility kernels

extern "C" __global__ void zero_counters_kernel(int* cnt0, int* cnt1) {
  if (threadIdx.x < NEXP) { cnt0[threadIdx.x] = 0; cnt1[threadIdx.x] = 0; }
}

extern "C" __global__ void convert_x_kernel(const float* __restrict__ x,
                                            uint16_t* __restrict__ xb, int n) {
  int i = blockIdx.x * blockDim.x + threadIdx.x;
  int stride = gridDim.x * blockDim.x;
  for (; i < n; i += stride) xb[i] = f2bf(x[i]);
}

// Tiled transpose+convert: src [E][M][N] fp32 -> dst [E][N][M] bf16.
// 64x64 tiles through LDS: coalesced reads AND coalesced writes.
#define TT 64
extern "C" __global__ void transpose_convert_kernel(const float* __restrict__ src,
                                                    uint16_t* __restrict__ dst,
                                                    int M, int N) {
  __shared__ uint16_t tile[TT][TT + 2];   // 33-dword row stride: conflict-free columns
  const int tilesN = N / TT;
  const int tilesPerE = (M / TT) * tilesN;
  const int bid = blockIdx.x;
  const int e = bid / tilesPerE;
  const int r = bid - e * tilesPerE;
  const int tm = r / tilesN;
  const int tn = r - tm * tilesN;
  const size_t per_e = (size_t)M * N;
  const float* s = src + (size_t)e * per_e + (size_t)(tm * TT) * N + tn * TT;
  uint16_t* d = dst + (size_t)e * per_e + (size_t)(tn * TT) * M + tm * TT;
  const int tid = threadIdx.x;
#pragma unroll
  for (int i = 0; i < (TT * TT) / 256; ++i) {
    int idx = tid + i * 256;
    int ml = idx >> 6, nl = idx & 63;
    tile[ml][nl] = f2bf(s[(size_t)ml * N + nl]);
  }
  __syncthreads();
#pragma unroll
  for (int i = 0; i < (TT * TT) / 256; ++i) {
    int idx = tid + i * 256;
    int nl = idx >> 6, ml = idx & 63;
    d[(size_t)nl * M + ml] = tile[ml][nl];
  }
}

// ---------------------------------------------------------------- gating

extern "C" __global__ void gate_kernel(const float* __restrict__ x,
                                       const float* __restrict__ wg,
                                       float* __restrict__ gates,
                                       int* cnt0, int* tok0, float* g0v,
                                       int* cnt1, int* tok1, float* g1v) {
  __shared__ float wg_s[HDIM * NEXP];   // 32 KB
  const int tid = threadIdx.x;
  for (int j = tid; j < HDIM * NEXP; j += 256) wg_s[j] = wg[j];
  __syncthreads();

  const int token = blockIdx.x * 8 + (tid >> 5);   // one wave32 per token
  const int lane = tid & 31;

  float acc[NEXP];
#pragma unroll
  for (int e = 0; e < NEXP; ++e) acc[e] = 0.f;

  const float* xr = x + (size_t)token * HDIM;
  for (int h = lane; h < HDIM; h += 32) {
    float xv = xr[h];
#pragma unroll
    for (int e = 0; e < NEXP; ++e) acc[e] += xv * wg_s[h * NEXP + e];
  }
#pragma unroll
  for (int e = 0; e < NEXP; ++e)
    for (int off = 16; off > 0; off >>= 1)
      acc[e] += __shfl_xor(acc[e], off, 32);

  if (lane == 0) {
    float m = acc[0];
#pragma unroll
    for (int e = 1; e < NEXP; ++e) m = fmaxf(m, acc[e]);
    float p[NEXP]; float s = 0.f;
#pragma unroll
    for (int e = 0; e < NEXP; ++e) { p[e] = expf(acc[e] - m); s += p[e]; }
#pragma unroll
    for (int e = 0; e < NEXP; ++e) p[e] /= s;

    int i0 = 0;
#pragma unroll
    for (int e = 1; e < NEXP; ++e) if (p[e] > p[i0]) i0 = e;
    int i1 = (i0 == 0) ? 1 : 0;
#pragma unroll
    for (int e = 0; e < NEXP; ++e) if (e != i0 && p[e] > p[i1]) i1 = e;

    const float den = p[i0] + p[i1];
    const float g0 = p[i0] / den;
    const float g1 = p[i1] / den;
#pragma unroll
    for (int e = 0; e < NEXP; ++e)
      gates[(size_t)token * NEXP + e] = (e == i0) ? g0 : ((e == i1) ? g1 : 0.f);

    int pos0 = atomicAdd(&cnt0[i0], 1);
    tok0[i0 * NTOK + pos0] = token; g0v[i0 * NTOK + pos0] = g0;
    int pos1 = atomicAdd(&cnt1[i1], 1);
    tok1[i1 * NTOK + pos1] = token; g1v[i1 * NTOK + pos1] = g1;
  }
}

// ---------------------------------------------------------------- expert GEMM (WMMA)

extern "C" __global__ void __launch_bounds__(256)
moe_expert_gemm(const uint16_t* __restrict__ xb,
                const uint16_t* __restrict__ w1t,   // [E][F][H] bf16
                const float*    __restrict__ b1,
                const uint16_t* __restrict__ w2t,   // [E][H][F] bf16
                const float*    __restrict__ b2,
                const int*      __restrict__ cnt,
                const int*      __restrict__ toks,
                const float*    __restrict__ gvals,
                float*          __restrict__ y,
                int accumulate) {
  const int e = blockIdx.y;
  const int tile = blockIdx.x;
  const int count = cnt[e];
  if (tile * TILE_M >= count) return;   // uniform per block

  extern __shared__ uint8_t smem[];
  uint16_t* xs = (uint16_t*)smem;                                     // 64 x 1032 bf16
  uint16_t* hs = (uint16_t*)(smem + TILE_M * XS_STRIDE * 2);          // 64 x 40 bf16
  int*   tok_s  = (int*)(smem + TILE_M * XS_STRIDE * 2 + TILE_M * HS_STRIDE * 2);
  float* gate_s = (float*)(tok_s + TILE_M);

  const int tid = threadIdx.x;
  if (tid < TILE_M) {
    int r = tile * TILE_M + tid;
    int rr = (r < count) ? r : (count - 1);
    tok_s[tid]  = toks[e * NTOK + rr];
    gate_s[tid] = gvals[e * NTOK + rr];
  }
  __syncthreads();

  // gather x tile into LDS with CDNA5 async loads (ASYNCcnt-tracked, no VGPR bounce)
  {
    const uint32_t xs_base = (uint32_t)(uintptr_t)(void*)xs;  // low 32 bits = LDS offset
    for (int j = tid; j < TILE_M * (HDIM * 2 / 16); j += 256) {   // 16B chunks
      int row = j >> 7;          // / 128 chunks per row
      int c16 = j & 127;
      const void* g = (const uint8_t*)(xb + (size_t)tok_s[row] * HDIM) + c16 * 16;
      uint32_t l = xs_base + (uint32_t)(row * (XS_STRIDE * 2) + c16 * 16);
      async_gather_b128(l, g);
    }
    wait_asynccnt0();
  }
  __syncthreads();

  const int wave = tid >> 5;
  const int lane = tid & 31;
  const int half = lane >> 4;
  const int l16  = lane & 15;

  const int s1_mr = wave >> 1;           // h-tile row block (tokens)
  const int s1_nc = (wave & 1) * 16;     // h-tile col offset within f-chunk
  const int s2_mr = wave >> 1;           // y row block
  const int s2_par = wave & 1;           // y col-tile parity

  const uint16_t* w1e = w1t + (size_t)e * FDIM * HDIM;
  const uint16_t* w2e = w2t + (size_t)e * HDIM * FDIM;
  const float* b1e = b1 + e * FDIM;
  const float* b2e = b2 + e * HDIM;

  v8f acc[32];
#pragma unroll
  for (int t = 0; t < 32; ++t) acc[t] = vzero8();

  const uint16_t* xrow = xs + (s1_mr * 16 + l16) * XS_STRIDE;
  const uint16_t* hrow = hs + (s2_mr * 16 + l16) * HS_STRIDE;

  for (int fc = 0; fc < FDIM / FCHUNK; ++fc) {
    const int fbase = fc * FCHUNK;

    // ---- stage 1: h(64x32) = relu(x_tile @ w1 + b1); one 16x16 tile per wave,
    //      software-pipelined so loads of step ks+1 overlap the WMMA of step ks.
    const uint16_t* bcol = w1e + (size_t)(fbase + s1_nc + l16) * HDIM + half * 16;
    if (fc + 1 < FDIM / FCHUNK)
      __builtin_prefetch(bcol + (size_t)FCHUNK * HDIM, 0, 0);   // global_prefetch_b8

    v8f c = vzero8();
    Frag a_cur = load_a_frag(xrow, half);
    Frag b_cur = load_b_frag(bcol);
#pragma unroll 4
    for (int ks = 0; ks < HDIM / 32 - 1; ++ks) {
      Frag a_nxt = load_a_frag(xrow + (ks + 1) * 32, half);
      Frag b_nxt = load_b_frag(bcol + (ks + 1) * 32);
      c = __builtin_amdgcn_wmma_f32_16x16x32_bf16(false, a_cur.f, false, b_cur.f,
                                                  (short)0, c, false, false);
      a_cur = a_nxt; b_cur = b_nxt;
    }
    c = __builtin_amdgcn_wmma_f32_16x16x32_bf16(false, a_cur.f, false, b_cur.f,
                                                (short)0, c, false, false);

    const float bias = b1e[fbase + s1_nc + l16];
#pragma unroll
    for (int i = 0; i < 8; ++i) {
      float v = c[i] + bias;
      v = v > 0.f ? v : 0.f;
      hs[(s1_mr * 16 + half * 8 + i) * HS_STRIDE + s1_nc + l16] = f2bf(v);
    }
    __syncthreads();

    // ---- stage 2: y_acc(64x1024) += h(64x32) @ w2[fchunk,:], pipelined B stream
    Frag ah = load_a_frag(hrow, half);
    const uint16_t* w2base = w2e + fbase + half * 16;
    Frag bw_cur = load_b_frag(w2base + (size_t)((s2_par) * 16 + l16) * FDIM);
#pragma unroll
    for (int t = 0; t < 31; ++t) {
      const int ct_n = (t + 1) * 2 + s2_par;
      Frag bw_nxt = load_b_frag(w2base + (size_t)(ct_n * 16 + l16) * FDIM);
      acc[t] = __builtin_amdgcn_wmma_f32_16x16x32_bf16(false, ah.f, false, bw_cur.f,
                                                       (short)0, acc[t], false, false);
      bw_cur = bw_nxt;
    }
    acc[31] = __builtin_amdgcn_wmma_f32_16x16x32_bf16(false, ah.f, false, bw_cur.f,
                                                      (short)0, acc[31], false, false);
    __syncthreads();
  }

  // ---- scatter y (slot0 writes, slot1 accumulates; each token unique per slot)
#pragma unroll
  for (int t = 0; t < 32; ++t) {
    const int col = (t * 2 + s2_par) * 16 + l16;
    const float bb = b2e[col];
#pragma unroll
    for (int i = 0; i < 8; ++i) {
      const int row = s2_mr * 16 + half * 8 + i;
      if (tile * TILE_M + row < count) {
        float val = gate_s[row] * (acc[t][i] + bb);
        float* dst = y + (size_t)tok_s[row] * HDIM + col;
        if (accumulate) *dst += val; else *dst = val;
      }
    }
  }
}

// ---------------------------------------------------------------- loss

extern "C" __global__ void loss_kernel(const float* __restrict__ gates,
                                       float* __restrict__ out_loss) {
  __shared__ float s_imp[256 * NEXP];
  __shared__ float s_load[256 * NEXP];
  const int tid = threadIdx.x;
  float imp[NEXP], ld[NEXP];
#pragma unroll
  for (int e = 0; e < NEXP; ++e) { imp[e] = 0.f; ld[e] = 0.f; }
  for (int t = tid; t < NTOK; t += 256) {
#pragma unroll
    for (int e = 0; e < NEXP; ++e) {
      float g = gates[(size_t)t * NEXP + e];
      imp[e] += g;
      ld[e]  += (g > 0.f) ? 1.f : 0.f;
    }
  }
#pragma unroll
  for (int e = 0; e < NEXP; ++e) { s_imp[tid * NEXP + e] = imp[e]; s_load[tid * NEXP + e] = ld[e]; }
  __syncthreads();
  for (int off = 128; off > 0; off >>= 1) {
    if (tid < off) {
#pragma unroll
      for (int e = 0; e < NEXP; ++e) {
        s_imp[tid * NEXP + e]  += s_imp[(tid + off) * NEXP + e];
        s_load[tid * NEXP + e] += s_load[(tid + off) * NEXP + e];
      }
    }
    __syncthreads();
  }
  if (tid == 0) {
    float mi = 0.f, ml = 0.f;
#pragma unroll
    for (int e = 0; e < NEXP; ++e) { mi += s_imp[e]; ml += s_load[e]; }
    mi /= NEXP; ml /= NEXP;
    float vi = 0.f, vl = 0.f;
#pragma unroll
    for (int e = 0; e < NEXP; ++e) {
      float d0 = s_imp[e] - mi;  vi += d0 * d0;
      float d1 = s_load[e] - ml; vl += d1 * d1;
    }
    vi /= (NEXP - 1); vl /= (NEXP - 1);
    float cv2i = vi / (mi * mi + 1e-10f);
    float cv2l = vl / (ml * ml + 1e-10f);
    *out_loss = (cv2i + cv2l) * 0.01f;
  }
}

// ---------------------------------------------------------------- launch

extern "C" void kernel_launch(void* const* d_in, const int* in_sizes, int n_in,
                              void* d_out, int out_size, void* d_ws, size_t ws_size,
                              hipStream_t stream) {
  (void)in_sizes; (void)n_in; (void)out_size; (void)ws_size;
  const float* x  = (const float*)d_in[0];
  const float* wg = (const float*)d_in[1];
  const float* w1 = (const float*)d_in[2];
  const float* b1 = (const float*)d_in[3];
  const float* w2 = (const float*)d_in[4];
  const float* b2 = (const float*)d_in[5];
  float* out = (float*)d_out;

  uint8_t* ws = (uint8_t*)d_ws;
  size_t off = 0;
  int* cnt0 = (int*)(ws + off);                 off += 128;
  int* cnt1 = (int*)(ws + off);                 off += 128;
  float* gates = (float*)(ws + off);            off += (size_t)NTOK * NEXP * 4;
  int*   tok0  = (int*)(ws + off);              off += (size_t)NEXP * NTOK * 4;
  float* g0v   = (float*)(ws + off);            off += (size_t)NEXP * NTOK * 4;
  int*   tok1  = (int*)(ws + off);              off += (size_t)NEXP * NTOK * 4;
  float* g1v   = (float*)(ws + off);            off += (size_t)NEXP * NTOK * 4;
  uint16_t* xb  = (uint16_t*)(ws + off);        off += (size_t)NTOK * HDIM * 2;
  uint16_t* w1t = (uint16_t*)(ws + off);        off += (size_t)NEXP * HDIM * FDIM * 2;
  uint16_t* w2t = (uint16_t*)(ws + off);        off += (size_t)NEXP * HDIM * FDIM * 2;

  zero_counters_kernel<<<1, 32, 0, stream>>>(cnt0, cnt1);
  convert_x_kernel<<<4096, 256, 0, stream>>>(x, xb, NTOK * HDIM);
  // w1 [E][H][F] -> w1t [E][F][H]; w2 [E][F][H] -> w2t [E][H][F]
  transpose_convert_kernel<<<NEXP * (HDIM / TT) * (FDIM / TT), 256, 0, stream>>>(w1, w1t, HDIM, FDIM);
  transpose_convert_kernel<<<NEXP * (FDIM / TT) * (HDIM / TT), 256, 0, stream>>>(w2, w2t, FDIM, HDIM);
  gate_kernel<<<NTOK / 8, 256, 0, stream>>>(x, wg, gates,
                                            cnt0, tok0, g0v, cnt1, tok1, g1v);

  const size_t smem_bytes = (size_t)TILE_M * XS_STRIDE * 2
                          + (size_t)TILE_M * HS_STRIDE * 2
                          + (size_t)TILE_M * 4 + (size_t)TILE_M * 4;
  dim3 ggrid(NTOK / TILE_M, NEXP);
  moe_expert_gemm<<<ggrid, 256, smem_bytes, stream>>>(xb, w1t, b1, w2t, b2,
                                                      cnt0, tok0, g0v, out, 0);
  moe_expert_gemm<<<ggrid, 256, smem_bytes, stream>>>(xb, w1t, b1, w2t, b2,
                                                      cnt1, tok1, g1v, out, 1);
  loss_kernel<<<1, 256, 0, stream>>>(gates, out + (size_t)NTOK * HDIM);
}
